// StochasticToucanTTS_72258529788608
// MI455X (gfx1250) — matrix operand order repaired and checked by reading, not compiled, not run on hardware
//
#include <hip/hip_runtime.h>
#include <hip/hip_bf16.h>
#include <math.h>

// ---------------------------------------------------------------------------
// StochasticToucanTTS forward, MI455X (gfx1250, wave32, WMMA).
// GEMMs: v_wmma_f32_16x16x32_f16; fp32 in memory, f16 tiles in LDS (A row-
// major, B transposed -> all fragments are ds_load_b128), double-buffered,
// 32x64 register tile per wave (8 WMMA per K-step, 0.75 KB LDS/WMMA).
// ---------------------------------------------------------------------------

constexpr int BSZ   = 8;
constexpr int TLEN  = 128;
constexpr int LLEN  = 1024;
constexpr int AD    = 192;
constexpr int NHEAD = 4;
constexpr int DHEAD = 48;   // AD / NHEAD
constexpr int HID   = 1536;
constexpr int ODM   = 80;
constexpr int UTTD  = 64;
constexpr int INFEAT= 62;
constexpr int CCH   = AD + UTTD; // flow conditioning channels = 256

typedef __attribute__((ext_vector_type(2)))  _Float16 v2h;
typedef __attribute__((ext_vector_type(4)))  _Float16 v4h;
typedef __attribute__((ext_vector_type(8)))  _Float16 v8h;
typedef __attribute__((ext_vector_type(16))) _Float16 v16h;
typedef __attribute__((ext_vector_type(8)))  float    v8f;

// ---------------------------- WMMA GEMM ------------------------------------
// C[M,N] = act(A[M,K] @ W[K,N] + bias). act: 0=none, 1=tanh, 2=silu
constexpr int TBM = 128;
constexpr int TBN = 64;
constexpr int TBK = 32;

__global__ void __launch_bounds__(128)
k_gemm(const float* __restrict__ A, const float* __restrict__ Wm,
       const float* __restrict__ bias, float* __restrict__ C,
       int M, int N, int K, int act)
{
  // Double-buffered tiles: A row-major [row][32], B transposed [col][32].
  __shared__ _Float16 sA [2][TBM * TBK];   // 2 x 8 KB
  __shared__ _Float16 sBt[2][TBN * TBK];   // 2 x 4 KB

  const int tid  = threadIdx.x;
  const int mb   = blockIdx.x * TBM;
  const int nb   = blockIdx.y * TBN;
  const int wave = tid >> 5;        // 0..3, each owns 32 rows
  const int lane = tid & 31;
  const int half = lane >> 4;
  const int lr   = lane & 15;

  auto stage = [&](int kt, int buf) {
    const int k0 = kt * TBK;
    // ---- A tile: fp32 -> f16, 4 elems/thread/iter ----
    const bool aInt = (mb + TBM <= M) && (k0 + TBK <= K) && ((K & 3) == 0);
    if (aInt) {
      for (int i = tid; i < (TBM * TBK) / 4; i += 128) {
        const int base = i * 4;
        const int r = base >> 5, c = base & 31;
        const float4 f = *(const float4*)&A[(size_t)(mb + r) * K + (k0 + c)];
        v4h o; o[0] = (_Float16)f.x; o[1] = (_Float16)f.y;
               o[2] = (_Float16)f.z; o[3] = (_Float16)f.w;
        *(v4h*)&sA[buf][base] = o;
      }
    } else {
      for (int i = tid; i < (TBM * TBK) / 4; i += 128) {
        const int base = i * 4;
        const int r = base >> 5, c = base & 31;
        const int gr = mb + r, gc = k0 + c;
        v4h o;
#pragma unroll
        for (int j = 0; j < 4; ++j) {
          const int g = gc + j;
          o[j] = (_Float16)((gr < M && g < K) ? A[(size_t)gr * K + g] : 0.0f);
        }
        *(v4h*)&sA[buf][base] = o;
      }
    }
    // ---- B tile: transpose into [col][k], K-pair/thread (b32 DS store) ----
    const bool bInt = (nb + TBN <= N) && (k0 + TBK <= K);
    if (bInt) {
      for (int i = tid; i < (TBK / 2) * TBN; i += 128) {
        const int c = i & 63, k = (i >> 6) * 2;
        const float v0 = Wm[(size_t)(k0 + k)     * N + nb + c];
        const float v1 = Wm[(size_t)(k0 + k + 1) * N + nb + c];
        v2h p; p[0] = (_Float16)v0; p[1] = (_Float16)v1;
        *(v2h*)&sBt[buf][c * TBK + k] = p;
      }
    } else {
      for (int i = tid; i < (TBK / 2) * TBN; i += 128) {
        const int c = i & 63, k = (i >> 6) * 2;
        const int gr = k0 + k, gc = nb + c;
        const float v0 = (gr < K && gc < N)     ? Wm[(size_t)gr * N + gc]       : 0.0f;
        const float v1 = (gr + 1 < K && gc < N) ? Wm[(size_t)(gr + 1) * N + gc] : 0.0f;
        v2h p; p[0] = (_Float16)v0; p[1] = (_Float16)v1;
        *(v2h*)&sBt[buf][c * TBK + k] = p;
      }
    }
  };

  v8f acc[2][4];
#pragma unroll
  for (int mi = 0; mi < 2; ++mi)
#pragma unroll
    for (int i = 0; i < 4; ++i)
#pragma unroll
      for (int j = 0; j < 8; ++j) acc[mi][i][j] = 0.0f;

  const int nk = (K + TBK - 1) / TBK;
  stage(0, 0);
  __syncthreads();

  for (int kt = 0; kt < nk; ++kt) {
    const int cur = kt & 1;
    if (kt + 1 < nk) stage(kt + 1, cur ^ 1);   // overlap VMEM with WMMA

    // Two A fragments (rows wave*32 .. +31); ISA 16-bit A 16x32 layout:
    // elems 0-7 = K[8h..8h+7], elems 8-15 = K[16+8h..23+8h] -> 2x b128 each.
    v16h af[2];
#pragma unroll
    for (int mi = 0; mi < 2; ++mi) {
      const v8h* pa = (const v8h*)&sA[cur][(wave * 32 + mi * 16 + lr) * TBK];
      af[mi] = __builtin_shufflevector(pa[half], pa[2 + half],
          0,1,2,3,4,5,6,7,8,9,10,11,12,13,14,15);
    }
#pragma unroll
    for (int nt = 0; nt < 4; ++nt) {
      // B fragment: elems 0-7 = K[8h..8h+7], elems 8-15 = K[16+8h..23+8h] at column nt*16+lr,
      // in the transposed tile -> 2x b128; reused by 2 WMMAs.
      const v8h* pb = (const v8h*)&sBt[cur][(nt * 16 + lr) * TBK];
      const v16h bf = __builtin_shufflevector(pb[half], pb[2 + half],
          0,1,2,3,4,5,6,7,8,9,10,11,12,13,14,15);
#pragma unroll
      for (int mi = 0; mi < 2; ++mi)
        acc[mi][nt] = __builtin_amdgcn_wmma_f32_16x16x32_f16(
            false, af[mi], false, bf, (short)0, acc[mi][nt], false, false);
    }
    __syncthreads();
  }

  // Epilogue: C layout -> row = wave*32 + mi*16 + 8*half + v, col = nt*16+lr.
#pragma unroll
  for (int mi = 0; mi < 2; ++mi)
#pragma unroll
    for (int nt = 0; nt < 4; ++nt) {
      const int col = nb + nt * 16 + lr;
#pragma unroll
      for (int vv = 0; vv < 8; ++vv) {
        const int row = mb + wave * 32 + mi * 16 + 8 * half + vv;
        if (row < M && col < N) {
          float x = acc[mi][nt][vv];
          if (bias) x += bias[col];
          if (act == 1)      x = tanhf(x);
          else if (act == 2) x = x * (1.0f / (1.0f + __expf(-x)));
          C[(size_t)row * N + col] = x;
        }
      }
    }
}

// ---------------------------- LayerNorm ------------------------------------
__global__ void __launch_bounds__(AD)
k_layernorm(const float* __restrict__ x, float* __restrict__ y,
            const float* __restrict__ g, const float* __restrict__ bta,
            const float* __restrict__ rowmask, int R)
{
  __shared__ float s[AD];
  const int row = blockIdx.x;
  const int c   = threadIdx.x;
  float val = x[(size_t)row * AD + c];
  s[c] = val;
  __syncthreads();
  if (c < 64) s[c] = s[c] + s[c + 64] + s[c + 128];
  __syncthreads();
  for (int st = 32; st > 0; st >>= 1) { if (c < st) s[c] += s[c + st]; __syncthreads(); }
  const float mean = s[0] * (1.0f / AD);
  __syncthreads();
  const float d = val - mean;
  s[c] = d * d;
  __syncthreads();
  if (c < 64) s[c] = s[c] + s[c + 64] + s[c + 128];
  __syncthreads();
  for (int st = 32; st > 0; st >>= 1) { if (c < st) s[c] += s[c + st]; __syncthreads(); }
  const float var = s[0] * (1.0f / AD);
  float o = d * rsqrtf(var + 1e-5f) * g[c] + bta[c];
  if (rowmask) o *= rowmask[row];
  y[(size_t)row * AD + c] = o;
}

// ---------------------------- elementwise ----------------------------------
__global__ void k_axpy(float* __restrict__ y, const float* __restrict__ x,
                       float s, size_t n)
{
  size_t i = (size_t)blockIdx.x * blockDim.x + threadIdx.x;
  if (i < n) y[i] += s * x[i];
}

__global__ void k_enc_prelude(float* __restrict__ h, const float* __restrict__ lang_emb,
                              const int* __restrict__ lang_ids,
                              const float* __restrict__ uttp,
                              const float* __restrict__ alpha)
{
  int idx = blockIdx.x * blockDim.x + threadIdx.x;
  if (idx >= BSZ * TLEN * AD) return;
  int c = idx % AD, bt = idx / AD, t = bt % TLEN, b = bt / TLEN;
  int j2 = (c >> 1) << 1;
  float freq = __expf(-logf(10000.0f) * (float)j2 / (float)AD);
  float arg  = (float)t * freq;
  float pe   = (c & 1) ? __cosf(arg) : __sinf(arg);
  h[idx] += lang_emb[(size_t)lang_ids[b] * AD + c] + alpha[0] * pe + uttp[b * AD + c];
}

__global__ void k_glu(const float* __restrict__ in, float* __restrict__ out, int R)
{
  int idx = blockIdx.x * blockDim.x + threadIdx.x;
  if (idx >= R * AD) return;
  int c = idx % AD, r = idx / AD;
  float a = in[(size_t)r * 2 * AD + c];
  float g = in[(size_t)r * 2 * AD + AD + c];
  out[idx] = a * (1.0f / (1.0f + __expf(-g)));
}

__global__ void k_dwconv_silu(const float* __restrict__ in, float* __restrict__ out,
                              const float* __restrict__ w, int kc, int Tseq)
{
  int idx = blockIdx.x * blockDim.x + threadIdx.x;
  if (idx >= BSZ * Tseq * AD) return;
  int c = idx % AD, bt = idx / AD, t = bt % Tseq, b = bt / Tseq;
  int pad = (kc - 1) >> 1;
  float s = 0.0f;
  for (int i = 0; i < kc; ++i) {
    int tt = t + i - pad;
    if (tt >= 0 && tt < Tseq)
      s += in[((size_t)b * Tseq + tt) * AD + c] * w[i * AD + c];
  }
  out[idx] = s * (1.0f / (1.0f + __expf(-s)));
}

__global__ void k_add_embed(float* __restrict__ h, const float* __restrict__ sc,
                            const float* __restrict__ w, const float* __restrict__ bb)
{
  int idx = blockIdx.x * blockDim.x + threadIdx.x;
  if (idx >= BSZ * TLEN * AD) return;
  int c = idx % AD, bt = idx / AD;
  h[idx] += sc[bt] * w[c] + bb[c];
}

__global__ void k_cond(const float* __restrict__ h, const float* __restrict__ utt,
                       float* __restrict__ cond)
{
  int idx = blockIdx.x * blockDim.x + threadIdx.x;
  if (idx >= BSZ * TLEN * CCH) return;
  int c = idx % CCH, bt = idx / CCH, b = bt / TLEN;
  cond[idx] = (c < AD) ? h[(size_t)bt * AD + c] : utt[b * UTTD + (c - AD)];
}

// ---------------------------- fused attention ------------------------------
// One wave (32 lanes) per (b, head, q). Scores staged in LDS, wave32
// shuffle reductions, no global score tensor (saves 134 MB in the decoder).
__global__ void __launch_bounds__(256)
k_attention(const float* __restrict__ q, const float* __restrict__ k,
            const float* __restrict__ v, const float* __restrict__ maskr,
            float* __restrict__ o, int Tseq)
{
  __shared__ float sc[8][LLEN + 8];
  const int wave = threadIdx.x >> 5, lane = threadIdx.x & 31;
  const int gidx = blockIdx.x * 8 + wave;        // grid sized exactly
  const int qi = gidx % Tseq;
  const int bh = gidx / Tseq;
  const int hh = bh % NHEAD;
  const int b  = bh / NHEAD;

  const float* qp = q + ((size_t)b * Tseq + qi) * AD + hh * DHEAD;
  float qreg[DHEAD];
#pragma unroll
  for (int d = 0; d < DHEAD; ++d) qreg[d] = qp[d];

  const float scale = rsqrtf((float)DHEAD);
  float mx = -3.0e38f;
  for (int kk = lane; kk < Tseq; kk += 32) {
    const float* kp = k + ((size_t)b * Tseq + kk) * AD + hh * DHEAD;
    float s = 0.0f;
#pragma unroll
    for (int d = 0; d < DHEAD; ++d) s += qreg[d] * kp[d];
    s *= scale;
    if (maskr[b * Tseq + kk] < 0.5f) s = -1.0e9f;
    sc[wave][kk] = s;
    mx = fmaxf(mx, s);
  }
  for (int off = 16; off > 0; off >>= 1) mx = fmaxf(mx, __shfl_xor(mx, off, 32));
  __syncthreads();

  float sum = 0.0f;
  for (int kk = lane; kk < Tseq; kk += 32) {
    float e = __expf(sc[wave][kk] - mx);
    sc[wave][kk] = e;
    sum += e;
  }
  for (int off = 16; off > 0; off >>= 1) sum += __shfl_xor(sum, off, 32);
  const float inv = 1.0f / sum;
  __syncthreads();

  float o0 = 0.0f, o1 = 0.0f;
  for (int kk = 0; kk < Tseq; ++kk) {
    float w = sc[wave][kk];
    const float* vp = v + ((size_t)b * Tseq + kk) * AD + hh * DHEAD;
    o0 += w * vp[lane];
    if (lane + 32 < DHEAD) o1 += w * vp[lane + 32];
  }
  float* op = o + ((size_t)b * Tseq + qi) * AD + hh * DHEAD;
  op[lane] = o0 * inv;
  if (lane + 32 < DHEAD) op[lane + 32] = o1 * inv;
}

// ---------------------------- flows & losses -------------------------------
struct FlowArgs { const float* W[6]; const float* bb[6]; int nsteps; int ksz; };

__global__ void k_flow_nll(const float* __restrict__ cond, const float* __restrict__ xin,
                           const int* __restrict__ dur, const float* __restrict__ tmaskf,
                           FlowArgs fa, float* __restrict__ acc2)
{
  int idx = blockIdx.x * blockDim.x + threadIdx.x;
  if (idx >= BSZ * TLEN) return;
  int t = idx % TLEN, b = idx / TLEN;
  float z  = dur ? logf((float)dur[idx] + 1.0f) : xin[idx];
  float ld = 0.0f;
  const int pad = (fa.ksz - 1) >> 1;
  for (int s = 0; s < fa.nsteps; ++s) {
    float st0 = fa.bb[s][0], st1 = fa.bb[s][1];
    for (int i = 0; i < fa.ksz; ++i) {
      int tt = t + i - pad;
      if (tt < 0 || tt >= TLEN) continue;
      const float* cp = cond + ((size_t)b * TLEN + tt) * CCH;
      const float* wp = fa.W[s] + (size_t)i * CCH * 2;
      for (int c = 0; c < CCH; ++c) {
        float cv = cp[c];
        st0 += cv * wp[c * 2 + 0];
        st1 += cv * wp[c * 2 + 1];
      }
    }
    float m = st0, lg = tanhf(st1);
    z  = (z - m) * __expf(-lg);
    ld += lg;
  }
  float nll = 0.5f * z * z + 0.9189385332046727f + ld;
  float mm  = tmaskf[idx];
  atomicAdd(&acc2[0], nll * mm);
  atomicAdd(&acc2[1], mm);
}

__global__ void k_tmask(const int* __restrict__ tl, float* __restrict__ tm)
{
  int idx = blockIdx.x * blockDim.x + threadIdx.x;
  if (idx >= BSZ * TLEN) return;
  tm[idx] = ((idx % TLEN) < tl[idx / TLEN]) ? 1.0f : 0.0f;
}

__global__ void k_cumsum(const int* __restrict__ durr, const float* __restrict__ tm,
                         int* __restrict__ cum)
{
  int b = threadIdx.x;
  if (b >= BSZ) return;
  int s = 0;
  for (int t = 0; t < TLEN; ++t) {
    s += (tm[b * TLEN + t] > 0.5f) ? durr[b * TLEN + t] : 0;
    cum[b * TLEN + t] = s;
  }
}

__global__ void k_fmask(const int* __restrict__ cum, const int* __restrict__ sl,
                        float* __restrict__ fm)
{
  int idx = blockIdx.x * blockDim.x + threadIdx.x;
  if (idx >= BSZ * LLEN) return;
  int b = idx / LLEN, l = idx % LLEN;
  int tot = cum[b * TLEN + TLEN - 1];
  int lim = sl[b] < tot ? sl[b] : tot;
  fm[idx] = (l < lim) ? 1.0f : 0.0f;
}

__global__ void k_regulate(const float* __restrict__ h, const int* __restrict__ cum,
                           const float* __restrict__ fm, float* __restrict__ x,
                           const float* __restrict__ alpha)
{
  int idx = blockIdx.x * blockDim.x + threadIdx.x;
  if (idx >= BSZ * LLEN * AD) return;
  int c = idx % AD, bl = idx / AD, l = bl % LLEN, b = bl / LLEN;
  const int* cb = cum + b * TLEN;
  int lo = 0, hi = TLEN;
  while (lo < hi) { int mid = (lo + hi) >> 1; if (cb[mid] <= l) lo = mid + 1; else hi = mid; }
  int fi = lo < TLEN - 1 ? lo : TLEN - 1;
  float val = h[((size_t)b * TLEN + fi) * AD + c] * fm[bl];
  int j2 = (c >> 1) << 1;
  float freq = __expf(-logf(10000.0f) * (float)j2 / (float)AD);
  float arg  = (float)l * freq;
  float pe   = (c & 1) ? __cosf(arg) : __sinf(arg);
  x[idx] = val + alpha[0] * pe;
}

__global__ void k_l1(const float* __restrict__ pred, const float* __restrict__ gold,
                     const float* __restrict__ fm, float* __restrict__ acc2)
{
  int idx = blockIdx.x * blockDim.x + threadIdx.x;
  if (idx >= BSZ * LLEN) return;
  float m = fm[idx];
  float s = 0.0f;
  const float* pp = pred + (size_t)idx * ODM;
  const float* gp = gold + (size_t)idx * ODM;
  for (int o = 0; o < ODM; ++o) s += fabsf(pp[o] - gp[o]);
  atomicAdd(&acc2[0], s * m);
  atomicAdd(&acc2[1], m);
}

__global__ void k_finalize(const float* __restrict__ acc, float* __restrict__ out)
{
  if (threadIdx.x == 0 && blockIdx.x == 0) {
    out[0] = acc[0] / (fmaxf(acc[1], 1.0f) * (float)ODM);  // l1
    out[1] = acc[2] / fmaxf(acc[3], 1.0f);                 // duration
    out[2] = acc[4] / fmaxf(acc[5], 1.0f);                 // pitch
    out[3] = acc[6] / fmaxf(acc[7], 1.0f);                 // energy
  }
}

// ---------------------------- host side ------------------------------------
struct BlockP {
  const float *dw, *ff1_b1, *ff1_b2, *ff1_w1, *ff1_w2,
              *ff2_b1, *ff2_b2, *ff2_w1, *ff2_w2,
              *ln_conv_b, *ln_conv_g, *ln_ff1_b, *ln_ff1_g,
              *ln_ff2_b, *ln_ff2_g, *ln_mha_b, *ln_mha_g,
              *ln_out_b, *ln_out_g, *pw1, *pw1_b, *pw2, *pw2_b,
              *wk, *wo, *wq, *wv;
};

static inline void gemm(hipStream_t st, const float* A, const float* Wm,
                        const float* bias, float* C, int M, int N, int K, int act)
{
  dim3 g((M + TBM - 1) / TBM, (N + TBN - 1) / TBN);
  k_gemm<<<g, 128, 0, st>>>(A, Wm, bias, C, M, N, K, act);
}

static void run_block(hipStream_t st, float* x, int Tseq, const float* maskr,
                      const BlockP& p, int kconv, float* hln, float* big,
                      float* qb, float* kb, float* vb, float* ab, float* tb)
{
  const int R = BSZ * Tseq;
  const size_t nx = (size_t)R * AD;
  const int eb = (int)((nx + 255) / 256);
  // FF1 (half-residual)
  k_layernorm<<<R, AD, 0, st>>>(x, hln, p.ln_ff1_g, p.ln_ff1_b, nullptr, R);
  gemm(st, hln, p.ff1_w1, p.ff1_b1, big, R, HID, AD, 2);
  gemm(st, big, p.ff1_w2, p.ff1_b2, tb, R, AD, HID, 0);
  k_axpy<<<eb, 256, 0, st>>>(x, tb, 0.5f, nx);
  // MHSA
  k_layernorm<<<R, AD, 0, st>>>(x, hln, p.ln_mha_g, p.ln_mha_b, nullptr, R);
  gemm(st, hln, p.wq, nullptr, qb, R, AD, AD, 0);
  gemm(st, hln, p.wk, nullptr, kb, R, AD, AD, 0);
  gemm(st, hln, p.wv, nullptr, vb, R, AD, AD, 0);
  k_attention<<<(BSZ * NHEAD * Tseq) / 8, 256, 0, st>>>(qb, kb, vb, maskr, ab, Tseq);
  gemm(st, ab, p.wo, nullptr, tb, R, AD, AD, 0);
  k_axpy<<<eb, 256, 0, st>>>(x, tb, 1.0f, nx);
  // Conv module
  k_layernorm<<<R, AD, 0, st>>>(x, hln, p.ln_conv_g, p.ln_conv_b, maskr, R);
  gemm(st, hln, p.pw1, p.pw1_b, big, R, 2 * AD, AD, 0);
  k_glu<<<eb, 256, 0, st>>>(big, tb, R);
  k_dwconv_silu<<<eb, 256, 0, st>>>(tb, ab, p.dw, kconv, Tseq);
  gemm(st, ab, p.pw2, p.pw2_b, tb, R, AD, AD, 0);
  k_axpy<<<eb, 256, 0, st>>>(x, tb, 1.0f, nx);
  // FF2 (half-residual)
  k_layernorm<<<R, AD, 0, st>>>(x, hln, p.ln_ff2_g, p.ln_ff2_b, nullptr, R);
  gemm(st, hln, p.ff2_w1, p.ff2_b1, big, R, HID, AD, 2);
  gemm(st, big, p.ff2_w2, p.ff2_b2, tb, R, AD, HID, 0);
  k_axpy<<<eb, 256, 0, st>>>(x, tb, 0.5f, nx);
  // Final LN (in-place safe: per-thread read before any write)
  k_layernorm<<<R, AD, 0, st>>>(x, x, p.ln_out_g, p.ln_out_b, nullptr, R);
}

extern "C" void kernel_launch(void* const* d_in, const int* in_sizes, int n_in,
                              void* d_out, int out_size, void* d_ws, size_t ws_size,
                              hipStream_t stream)
{
  (void)in_sizes; (void)n_in; (void)out_size; (void)ws_size;

  // ---- inputs (setup_inputs dict order) ----
  const float* text        = (const float*)d_in[0];
  const int*   text_len    = (const int*)  d_in[1];
  const float* gold_speech = (const float*)d_in[2];
  const int*   speech_len  = (const int*)  d_in[3];
  const int*   gold_dur    = (const int*)  d_in[4];
  const float* gold_pitch  = (const float*)d_in[5];
  const float* gold_energy = (const float*)d_in[6];
  const float* utt         = (const float*)d_in[7];
  const int*   lang_ids    = (const int*)  d_in[8];

  // ---- params: jax pytree leaves (dict keys sorted, lists in order) ----
  int ip = 9;
  auto F = [&]() -> const float* { return (const float*)d_in[ip++]; };
  auto loadBlock = [&](BlockP& bp) {
    bp.dw = F(); bp.ff1_b1 = F(); bp.ff1_b2 = F(); bp.ff1_w1 = F(); bp.ff1_w2 = F();
    bp.ff2_b1 = F(); bp.ff2_b2 = F(); bp.ff2_w1 = F(); bp.ff2_w2 = F();
    bp.ln_conv_b = F(); bp.ln_conv_g = F(); bp.ln_ff1_b = F(); bp.ln_ff1_g = F();
    bp.ln_ff2_b = F(); bp.ln_ff2_g = F(); bp.ln_mha_b = F(); bp.ln_mha_g = F();
    bp.ln_out_b = F(); bp.ln_out_g = F(); bp.pw1 = F(); bp.pw1_b = F();
    bp.pw2 = F(); bp.pw2_b = F(); bp.wk = F(); bp.wo = F(); bp.wq = F(); bp.wv = F();
  };

  const float* alpha_dec = F();
  const float* alpha_enc = F();
  BlockP dec[6]; for (int i = 0; i < 6; ++i) loadBlock(dec[i]);
  FlowArgs durF{}; durF.nsteps = 6; durF.ksz = 5;
  for (int i = 0; i < 6; ++i) { durF.W[i] = F(); durF.bb[i] = F(); }
  const float* emb_b1 = F(); const float* emb_b2 = F();
  const float* emb_w1 = F(); const float* emb_w2 = F();
  BlockP enc[6]; for (int i = 0; i < 6; ++i) loadBlock(enc[i]);
  const float* energy_emb_b = F(); const float* energy_emb_w = F();
  FlowArgs enF{}; enF.nsteps = 3; enF.ksz = 3;
  for (int i = 0; i < 3; ++i) { enF.W[i] = F(); enF.bb[i] = F(); }
  const float* feat_b = F(); const float* feat_w = F();
  const float* lang_emb = F();
  const float* pitch_emb_b = F(); const float* pitch_emb_w = F();
  FlowArgs piF{}; piF.nsteps = 6; piF.ksz = 5;
  for (int i = 0; i < 6; ++i) { piF.W[i] = F(); piF.bb[i] = F(); }
  const float* utt_b = F(); const float* utt_w = F();

  // ---- workspace carve-up (floats) ----
  float* WS = (float*)d_ws;
  size_t off = 0;
  auto alloc = [&](size_t n) { float* p = WS + off; off += n; return p; };
  float* acc    = alloc(16);
  float* tmaskf = alloc((size_t)BSZ * TLEN);
  float* fmaskf = alloc((size_t)BSZ * LLEN);
  int*   cum    = (int*)alloc((size_t)BSZ * TLEN);
  float* h      = alloc((size_t)BSZ * TLEN * AD);
  float* t100   = alloc((size_t)BSZ * TLEN * 100);
  float* uttp   = alloc((size_t)BSZ * AD);
  float* cond   = alloc((size_t)BSZ * TLEN * CCH);
  float* x      = alloc((size_t)BSZ * LLEN * AD);
  float* hln    = alloc((size_t)BSZ * LLEN * AD);
  float* big    = alloc((size_t)BSZ * LLEN * HID);
  float* qb     = alloc((size_t)BSZ * LLEN * AD);
  float* kb     = alloc((size_t)BSZ * LLEN * AD);
  float* vb     = alloc((size_t)BSZ * LLEN * AD);
  float* ab     = alloc((size_t)BSZ * LLEN * AD);
  float* tb     = alloc((size_t)BSZ * LLEN * AD);

  hipMemsetAsync(acc, 0, 16 * sizeof(float), stream);

  const int NT  = BSZ * TLEN;            // 1024 encoder rows
  const int g1t = (NT + 255) / 256;
  const int ght = (NT * AD + 255) / 256;

  // Masks
  k_tmask<<<g1t, 256, 0, stream>>>(text_len, tmaskf);

  // Text embedding MLP + additive conditioning
  gemm(stream, text, emb_w1, emb_b1, t100, NT, 100, INFEAT, 1);
  gemm(stream, t100, emb_w2, emb_b2, h, NT, AD, 100, 0);
  gemm(stream, utt, utt_w, utt_b, uttp, BSZ, AD, UTTD, 0);
  k_enc_prelude<<<ght, 256, 0, stream>>>(h, lang_emb, lang_ids, uttp, alpha_enc);

  // Encoder
  for (int i = 0; i < 6; ++i)
    run_block(stream, h, TLEN, tmaskf, enc[i], 7, hln, big, qb, kb, vb, ab, tb);

  const int gcond = (NT * CCH + 255) / 256;
  // Pitch flow NLL, then pitch embedding
  k_cond<<<gcond, 256, 0, stream>>>(h, utt, cond);
  k_flow_nll<<<g1t, 256, 0, stream>>>(cond, gold_pitch, nullptr, tmaskf, piF, acc + 4);
  k_add_embed<<<ght, 256, 0, stream>>>(h, gold_pitch, pitch_emb_w, pitch_emb_b);
  // Energy flow NLL, then energy embedding
  k_cond<<<gcond, 256, 0, stream>>>(h, utt, cond);
  k_flow_nll<<<g1t, 256, 0, stream>>>(cond, gold_energy, nullptr, tmaskf, enF, acc + 6);
  k_add_embed<<<ght, 256, 0, stream>>>(h, gold_energy, energy_emb_w, energy_emb_b);
  // Duration flow NLL
  k_cond<<<gcond, 256, 0, stream>>>(h, utt, cond);
  k_flow_nll<<<g1t, 256, 0, stream>>>(cond, nullptr, gold_dur, tmaskf, durF, acc + 2);

  // Length regulation -> decoder input
  k_cumsum<<<1, 32, 0, stream>>>(gold_dur, tmaskf, cum);
  k_fmask<<<(BSZ * LLEN + 255) / 256, 256, 0, stream>>>(cum, speech_len, fmaskf);
  k_regulate<<<(BSZ * LLEN * AD + 255) / 256, 256, 0, stream>>>(h, cum, fmaskf, x, alpha_dec);

  // Decoder
  for (int i = 0; i < 6; ++i)
    run_block(stream, x, LLEN, fmaskf, dec[i], 31, hln, big, qb, kb, vb, ab, tb);

  // Output projection straight into d_out (+4 for the loss stack)
  float* outs = (float*)d_out;
  gemm(stream, x, feat_w, feat_b, outs + 4, BSZ * LLEN, ODM, AD, 0);

  // Losses
  k_l1<<<(BSZ * LLEN + 255) / 256, 256, 0, stream>>>(outs + 4, gold_speech, fmaskf, acc);
  k_finalize<<<1, 32, 0, stream>>>(acc, outs);
}
